// EfficientAttention_18159121727550
// MI455X (gfx1250) — matrix operand order repaired and checked by reading, not compiled, not run on hardware
//
#include <hip/hip_runtime.h>
#include <hip/hip_bf16.h>

typedef __bf16 bf16_t;
typedef __attribute__((ext_vector_type(16))) __bf16 v16bf;
typedef __attribute__((ext_vector_type(8)))  __bf16 v8bf;
typedef __attribute__((ext_vector_type(8)))  float  v8f;

#define TILE_M 128
#define TILE_N 128
#define TILE_K 32
#define LDS_S  40   // padded LDS stride (bf16 elems), multiple of 8 -> 16B aligned rows

// ---------------------------------------------------------------------------
// Generic batched GEMM:  C[b] = alpha * A[b] @ B[b] + bias
//   A : bf16 row-major [M x K]            (lda, batch stride sA)
//   Bt: bf16 N-MAJOR   [N x K]  (= B^T)   (ldbt, batch stride sB)
//   outF (f32) and/or outBF (bf16): [M x N] (ldc, batch stride sC)
// Requires K % 32 == 0 (call sites: 512, 8192, 64).
// GUARD=false requires M % 128 == 0 && N % 128 == 0.
// ---------------------------------------------------------------------------
template <bool GUARD>
__global__ __launch_bounds__(256)
void gemm_bf16_wmma(const bf16_t* __restrict__ A, long long lda, long long sA,
                    const bf16_t* __restrict__ Bt, long long ldbt, long long sB,
                    float* __restrict__ outF, bf16_t* __restrict__ outBF,
                    long long ldc, long long sC,
                    int M, int N, int K,
                    const float* __restrict__ bias, float alpha)
{
  __shared__ bf16_t As[2][TILE_M * LDS_S];       // As[buf][m][k]
  __shared__ bf16_t Bs[2][TILE_N * LDS_S];       // Bs[buf][n][k] (K-contiguous)

  const int tid  = threadIdx.x;
  const int lane = tid & 31;
  const int wave = tid >> 5;
  const int wrow = wave & 3;                     // 4 waves x 32 rows
  const int wcol = wave >> 2;                    // 2 waves x 64 cols
  const int bM   = blockIdx.y * TILE_M;
  const int bN   = blockIdx.x * TILE_N;

  A  += (long long)blockIdx.z * sA;
  Bt += (long long)blockIdx.z * sB;

  const int l16   = lane & 15;
  const int lhalf = lane >> 4;                   // 0: lanes 0-15, 1: lanes 16-31
  const int a_kb  = lhalf * 8;                   // A frag: K = {kb..kb+7, kb+16..kb+23}
  const int b_kb  = lhalf * 8;                   // B frag: K = {kb..kb+7, kb+16..kb+23}

  v8f acc[2][4];
  const v8f vzero = {0.f,0.f,0.f,0.f,0.f,0.f,0.f,0.f};
#pragma unroll
  for (int i = 0; i < 2; ++i)
#pragma unroll
    for (int j = 0; j < 4; ++j) acc[i][j] = vzero;

  // symmetric vectorized staging of A and Bt tiles (16B loads + 16B LDS stores)
  auto stage = [&](int kt, int buf) {
#pragma unroll
    for (int i = 0; i < 2; ++i) {
      int idx = tid + i * 256;                   // 0..511
      int row = idx >> 2;                        // 0..127
      int grp = idx & 3;                         // 8-elem K group
      uint4 av = {0u,0u,0u,0u}, bv = {0u,0u,0u,0u};
      if (!GUARD || (bM + row) < M)
        av = *(const uint4*)(A + (long long)(bM + row) * lda + kt + grp * 8);
      if (!GUARD || (bN + row) < N)
        bv = *(const uint4*)(Bt + (long long)(bN + row) * ldbt + kt + grp * 8);
      *(uint4*)&As[buf][row * LDS_S + grp * 8] = av;
      *(uint4*)&Bs[buf][row * LDS_S + grp * 8] = bv;
    }
  };

  stage(0, 0);
  __syncthreads();

  const int nk = K / TILE_K;
  for (int t = 0; t < nk; ++t) {
    const int buf = t & 1;
    if (t + 1 < nk) {
      stage((t + 1) * TILE_K, buf ^ 1);          // overlap next tile with WMMAs
      __builtin_prefetch(A + (long long)(bM + (tid & 127)) * lda + (t + 2) * TILE_K, 0, 0);
      __builtin_prefetch(Bt + (long long)(bN + (tid & 127)) * ldbt + (t + 2) * TILE_K, 0, 0);
    }

    // fragments per CDNA5 16-bit WMMA lane layouts
    v16bf afrag[2], bfrag[4];
#pragma unroll
    for (int mt = 0; mt < 2; ++mt) {
      int m = wrow * 32 + mt * 16 + l16;
      v8bf lo = *(const v8bf*)&As[buf][m * LDS_S + a_kb];
      v8bf hi = *(const v8bf*)&As[buf][m * LDS_S + a_kb + 16];
      afrag[mt] = __builtin_shufflevector(lo, hi, 0,1,2,3,4,5,6,7,8,9,10,11,12,13,14,15);
    }
#pragma unroll
    for (int nt = 0; nt < 4; ++nt) {
      int n = wcol * 64 + nt * 16 + l16;
      v8bf blo = *(const v8bf*)&Bs[buf][n * LDS_S + b_kb];
      v8bf bhi = *(const v8bf*)&Bs[buf][n * LDS_S + b_kb + 16];
      bfrag[nt] = __builtin_shufflevector(blo, bhi, 0,1,2,3,4,5,6,7,8,9,10,11,12,13,14,15);
    }

#pragma unroll
    for (int mt = 0; mt < 2; ++mt)
#pragma unroll
      for (int nt = 0; nt < 4; ++nt)
        acc[mt][nt] = __builtin_amdgcn_wmma_f32_16x16x32_bf16(
            false, afrag[mt], false, bfrag[nt], (short)0, acc[mt][nt], false, false);

    __syncthreads();                             // single barrier per K-step
  }

  // epilogue: C/D layout -> VGPR r: M = r + 8*lhalf, N = l16
  const long long cbase = (long long)blockIdx.z * sC;
#pragma unroll
  for (int mt = 0; mt < 2; ++mt) {
#pragma unroll
    for (int nt = 0; nt < 4; ++nt) {
      int gn = bN + wcol * 64 + nt * 16 + l16;
      if (GUARD && gn >= N) continue;
      float bval = bias ? bias[gn] : 0.0f;
#pragma unroll
      for (int r = 0; r < 8; ++r) {
        int gm = bM + wrow * 32 + mt * 16 + r + lhalf * 8;
        if (GUARD && gm >= M) continue;
        float v = acc[mt][nt][r] * alpha + bval;
        long long off = cbase + (long long)gm * ldc + gn;
        if (outF)  outF[off]  = v;
        if (outBF) outBF[off] = (bf16_t)v;
      }
    }
  }
}

// ---------------------------------------------------------------------------
// Q softmax: per (token, head) over 64 head channels; one wave each.
// ---------------------------------------------------------------------------
__global__ __launch_bounds__(256)
void qsoftmax_kernel(const float* __restrict__ KQV, bf16_t* __restrict__ q_sm)
{
  int w    = blockIdx.x * 8 + (threadIdx.x >> 5);
  int lane = threadIdx.x & 31;
  int n = w >> 3;
  int h = w & 7;
  const float* row = KQV + (long long)n * 1536 + 512 + h * 64;
  float v0 = row[lane * 2], v1 = row[lane * 2 + 1];
  float m = fmaxf(v0, v1);
#pragma unroll
  for (int off = 16; off > 0; off >>= 1) m = fmaxf(m, __shfl_xor(m, off, 32));
  float e0 = __expf(v0 - m), e1 = __expf(v1 - m);
  float s = e0 + e1;
#pragma unroll
  for (int off = 16; off > 0; off >>= 1) s += __shfl_xor(s, off, 32);
  float inv = 1.0f / s;
  bf16_t* out = q_sm + (long long)n * 512 + h * 64 + lane * 2;
  out[0] = (bf16_t)(e0 * inv);
  out[1] = (bf16_t)(e1 * inv);
}

// ---------------------------------------------------------------------------
// K softmax over 8192 tokens per channel; writes BOTH layouts:
//   k_smT [512 x 8192] (Bt for context GEMM), k_sm [8192 x 512] (Bt for scores)
// ---------------------------------------------------------------------------
__global__ __launch_bounds__(256)
void ksoftmax_kernel(const float* __restrict__ KQV,
                     bf16_t* __restrict__ k_smT, bf16_t* __restrict__ k_sm)
{
  int c   = blockIdx.x;
  int tid = threadIdx.x;
  __shared__ float red[256];
  float v[32];
  float m = -INFINITY;
#pragma unroll
  for (int i = 0; i < 32; ++i) {
    v[i] = KQV[(long long)(i * 256 + tid) * 1536 + c];
    m = fmaxf(m, v[i]);
  }
  red[tid] = m; __syncthreads();
  for (int s = 128; s > 0; s >>= 1) {
    if (tid < s) red[tid] = fmaxf(red[tid], red[tid + s]);
    __syncthreads();
  }
  m = red[0]; __syncthreads();

  float sum = 0.0f;
#pragma unroll
  for (int i = 0; i < 32; ++i) { v[i] = __expf(v[i] - m); sum += v[i]; }
  red[tid] = sum; __syncthreads();
  for (int s = 128; s > 0; s >>= 1) {
    if (tid < s) red[tid] += red[tid + s];
    __syncthreads();
  }
  float inv = 1.0f / red[0];
#pragma unroll
  for (int i = 0; i < 32; ++i) {
    int n = i * 256 + tid;
    bf16_t b = (bf16_t)(v[i] * inv);
    k_smT[(long long)c * 8192 + n] = b;     // coalesced
    k_sm [(long long)n * 512  + c] = b;     // strided (small: 8 MB once)
  }
}

// ---------------------------------------------------------------------------
// values slice transpose+convert: VbfT[c][n] = (bf16) KQV[n][1024+c]
// ---------------------------------------------------------------------------
__global__ __launch_bounds__(256)
void vtranspose_kernel(const float* __restrict__ KQV, bf16_t* __restrict__ VbfT)
{
  int c   = blockIdx.x;
  int tid = threadIdx.x;
#pragma unroll
  for (int i = 0; i < 32; ++i) {
    int n = i * 256 + tid;
    VbfT[(long long)c * 8192 + n] = (bf16_t)KQV[(long long)n * 1536 + 1024 + c];
  }
}

// ---------------------------------------------------------------------------
// Packing / conversion helpers (all emit N-major / transposed bf16 weights)
// ---------------------------------------------------------------------------
__global__ void cvt_bf16_kernel(const float* __restrict__ src,
                                bf16_t* __restrict__ dst, long long count)
{
  long long i = (long long)blockIdx.x * 256 + threadIdx.x;
  if (i < count) dst[i] = (bf16_t)src[i];
}

// WcatT [1536 x 512]: row cout = output channel of [Wk|Wq|Wv], col cin
__global__ void pack_wT_kernel(const float* __restrict__ Wk,
                               const float* __restrict__ Wq,
                               const float* __restrict__ Wv,
                               bf16_t* __restrict__ WcatT)
{
  long long i = (long long)blockIdx.x * 256 + threadIdx.x;
  if (i >= 1536LL * 512LL) return;
  int cin  = (int)(i & 511);
  int cout = (int)(i >> 9);
  float v = (cout < 512)  ? Wk[(long long)cin * 512 + cout]
          : (cout < 1024) ? Wq[(long long)cin * 512 + (cout - 512)]
                          : Wv[(long long)cin * 512 + (cout - 1024)];
  WcatT[i] = (bf16_t)v;
}

// WrT [512 x 512]: WrT[cout][cin] = Wr[cin][cout]
__global__ void pack_wrT_kernel(const float* __restrict__ Wr,
                                bf16_t* __restrict__ WrT)
{
  long long i = (long long)blockIdx.x * 256 + threadIdx.x;
  if (i >= 512LL * 512LL) return;
  int cin  = (int)(i & 511);
  int cout = (int)(i >> 9);
  WrT[i] = (bf16_t)Wr[(long long)cin * 512 + cout];
}

__global__ void pack_b_kernel(const float* __restrict__ bk,
                              const float* __restrict__ bq,
                              const float* __restrict__ bv,
                              float* __restrict__ bcat)
{
  int c = blockIdx.x * 256 + threadIdx.x;
  if (c >= 1536) return;
  bcat[c] = (c < 512) ? bk[c] : (c < 1024) ? bq[c - 512] : bv[c - 1024];
}

// ---------------------------------------------------------------------------
extern "C" void kernel_launch(void* const* d_in, const int* in_sizes, int n_in,
                              void* d_out, int out_size, void* d_ws, size_t ws_size,
                              hipStream_t stream)
{
  (void)in_sizes; (void)n_in; (void)out_size; (void)ws_size;
  const int N = 8192, C = 512, H = 8, CK3 = 1536;

  const float* X  = (const float*)d_in[0];
  const float* Wk = (const float*)d_in[1];
  const float* bk = (const float*)d_in[2];
  const float* Wq = (const float*)d_in[3];
  const float* bq = (const float*)d_in[4];
  const float* Wv = (const float*)d_in[5];
  const float* bv = (const float*)d_in[6];
  const float* Wr = (const float*)d_in[7];
  const float* br = (const float*)d_in[8];

  float* attention = (float*)d_out;                       // [8192 x 512]
  float* scores    = (float*)d_out + (long long)N * C;    // [8192 x 8192]

  char* ws = (char*)d_ws;
  auto carve = [&](size_t bytes) {
    char* p = ws; ws += (bytes + 255) & ~size_t(255); return p;
  };
  bf16_t* Xbf   = (bf16_t*)carve((size_t)N * C * 2);
  bf16_t* WcatT = (bf16_t*)carve((size_t)CK3 * C * 2);
  float*  bcat  = (float*) carve((size_t)CK3 * 4);
  bf16_t* WrT   = (bf16_t*)carve((size_t)C * C * 2);
  float*  KQV   = (float*) carve((size_t)N * CK3 * 4);       // keys|queries|values f32
  bf16_t* k_smT = (bf16_t*)carve((size_t)C * N * 2);         // [512 x 8192]
  bf16_t* k_sm  = (bf16_t*)carve((size_t)N * C * 2);         // [8192 x 512]
  bf16_t* q_sm  = (bf16_t*)carve((size_t)N * C * 2);         // [8192 x 512]
  bf16_t* VbfT  = (bf16_t*)carve((size_t)C * N * 2);         // [512 x 8192]
  bf16_t* ctxT  = (bf16_t*)carve((size_t)H * 64 * 64 * 2);   // ctx^T per head
  bf16_t* agg   = (bf16_t*)carve((size_t)N * C * 2);

  // 1) conversions / packing (all weight operands produced N-major)
  cvt_bf16_kernel<<<(N * C + 255) / 256, 256, 0, stream>>>(X, Xbf, (long long)N * C);
  pack_wT_kernel<<<(CK3 * C + 255) / 256, 256, 0, stream>>>(Wk, Wq, Wv, WcatT);
  pack_b_kernel<<<(CK3 + 255) / 256, 256, 0, stream>>>(bk, bq, bv, bcat);
  pack_wrT_kernel<<<(C * C + 255) / 256, 256, 0, stream>>>(Wr, WrT);

  // 2) fused QKV projection (branchless): [8192x512] @ [512x1536] + bcat -> KQV f32
  gemm_bf16_wmma<false><<<dim3(CK3 / TILE_N, N / TILE_M, 1), 256, 0, stream>>>(
      Xbf, C, 0, WcatT, C, 0, KQV, nullptr, CK3, 0, N, CK3, C, bcat, 1.0f);

  // 3) softmaxes + values transpose
  ksoftmax_kernel<<<C, 256, 0, stream>>>(KQV, k_smT, k_sm);
  qsoftmax_kernel<<<N, 256, 0, stream>>>(KQV, q_sm);
  vtranspose_kernel<<<C, 256, 0, stream>>>(KQV, VbfT);

  // 4) ctxT[h] = v[h]^T @ k_sm[h]  ([64x8192]@[8192x64], Bt = k_smT rows)
  gemm_bf16_wmma<true><<<dim3(1, 1, H), 256, 0, stream>>>(
      VbfT, N, 64LL * N, k_smT, N, 64LL * N, nullptr, ctxT, 64, 64LL * 64,
      64, 64, N, nullptr, 1.0f);

  // 5) attended[h] = q_sm[h] @ ctx[h]  (Bt = ctxT) -> agg bf16
  gemm_bf16_wmma<true><<<dim3(1, N / TILE_M, H), 256, 0, stream>>>(
      q_sm, C, 64, ctxT, 64, 64LL * 64, nullptr, agg, C, 64,
      N, 64, 64, nullptr, 1.0f);

  // 6) attention = agg @ Wr + br (branchless; Bt = WrT) -> d_out f32
  gemm_bf16_wmma<false><<<dim3(C / TILE_N, N / TILE_M, 1), 256, 0, stream>>>(
      agg, C, 0, WrT, C, 0, attention, nullptr, C, 0, N, C, C, br, 1.0f);

  // 7) scores = (q_sm @ k_sm^T) / H (branchless; Bt = k_sm) -> [8192x8192] f32
  gemm_bf16_wmma<false><<<dim3(N / TILE_N, N / TILE_M, 1), 256, 0, stream>>>(
      q_sm, C, 0, k_sm, C, 0, scores, nullptr, N, 0, N, N, C, nullptr, 1.0f / H);
}
